// VQVAE_9036611190898
// MI455X (gfx1250) — compile-verified, matrix operand order checked by reading, not run on hardware
//
#include <hip/hip_runtime.h>
#include <hip/hip_bf16.h>
#include <stddef.h>
#include <stdint.h>

typedef __bf16 bf16_t;
typedef __attribute__((ext_vector_type(16))) __bf16 v16bf;
typedef __attribute__((ext_vector_type(8)))  float  v8f;
typedef __attribute__((ext_vector_type(4)))  unsigned int v4u;
typedef __attribute__((ext_vector_type(8)))  int v8i;
typedef __attribute__((ext_vector_type(4)))  int v4i;

union AFrag { v16bf v; __bf16 e[16]; };
union CFrag { v8f   v; float  e[8];  };

#define TB 256

// ---------------------------------------------------------------------------
// Tensor Data Mover: async 2D bf16 tile (rows x cols, row stride in elements)
// global -> LDS, with HW row padding: LDS row stride = cols*2B + (amount+1)*4B.
// Tracked by TENSORcnt.
// ---------------------------------------------------------------------------
__device__ __forceinline__ void tdm_load_2d_bf16(
    uint32_t lds_addr, const bf16_t* gptr, uint32_t rows, uint32_t cols,
    uint32_t stride_elems, uint32_t pad_interval, uint32_t pad_amount)
{
  uint64_t ga = (uint64_t)(uintptr_t)gptr;
  v4u g0;
  g0[0] = 1u;                                   // count=1, user descriptor
  g0[1] = lds_addr;                             // LDS byte address
  g0[2] = (uint32_t)ga;                         // global_addr[31:0]
  g0[3] = (uint32_t)((ga >> 32) & 0x1FFFFFFu) | (2u << 30);  // [56:32] | type=2
  v8i g1;
  uint32_t w0 = (1u << 16)                      // data_size = 2 bytes
              | (1u << 20)                      // pad_enable
              | (pad_interval << 22) | (pad_amount << 25);
  g1[0] = (int)w0;
  g1[1] = (int)((stride_elems & 0xFFFFu) << 16);                 // tensor_dim0[15:0]
  g1[2] = (int)(((stride_elems >> 16) & 0xFFFFu) | ((rows & 0xFFFFu) << 16)); // dim0 hi | dim1 lo
  g1[3] = (int)(((rows >> 16) & 0xFFFFu) | (cols << 16));        // dim1 hi | tile_dim0
  g1[4] = (int)rows;                                             // tile_dim1
  g1[5] = (int)stride_elems;                                     // tensor_dim0_stride[31:0]
  g1[6] = 0; g1[7] = 0;
  v4i gz = {0, 0, 0, 0};
#if __clang_major__ >= 23
  v8i gz8 = {0, 0, 0, 0, 0, 0, 0, 0};
  __builtin_amdgcn_tensor_load_to_lds(g0, g1, gz, gz, gz8, 0);
#else
  __builtin_amdgcn_tensor_load_to_lds(g0, g1, gz, gz, 0);
#endif
}

// Swizzle: k-index within 32-k step -> position so each lane's 16 halves of a
// WMMA operand are one contiguous 32B block: p = h*16 + g*8 + b
// (kk = g*16 + h*8 + b; h = lane-half, g = k-group).
__device__ __forceinline__ int wmma_perm(int kk) {
  return (((kk >> 3) & 1) << 4) + (((kk >> 4) & 1) << 3) + (kk & 7);
}

// ---------------------------------------------------------------------------
// Implicit-GEMM convolution (NHWC f32 activations, packed bf16 weights).
// DIL==2 => stride-1 conv over a virtual zero-stuffed input (ConvTranspose).
// Tile 128x128x32; 8 waves (2x4); wave = 4x2 of v_wmma_f32_16x16x32_bf16.
// B: pre-swizzled fragment-order weights copied by TDM (double-buffered,
// TENSORcnt); A: global->reg prefetch overlapped with WMMA, stored
// pre-swizzled. All operand fragments are single 32B LDS vector loads.
// ---------------------------------------------------------------------------
template<int CIN, int KH, int KW, int STRIDE, int PAD, int DIL>
__global__ __launch_bounds__(TB) void conv_gemm(
    const float* __restrict__ in, const bf16_t* __restrict__ Bw,
    const float* __restrict__ bias, float* __restrict__ out,
    int Hin, int Win, int Hout, int Wout, int Cout,
    int Kpad, int Npad, int relu_in)
{
  constexpr int KREAL = KH * KW * CIN;
  __shared__ __align__(64) __bf16 As[128][48];        // swizzled A, 32 k/step
  __shared__ __align__(64) __bf16 Bsw[2][128][48];    // swizzled B TDM image

  const int tid  = threadIdx.x;
  const int m0   = blockIdx.x * 128;
  const int n0   = blockIdx.y * 128;
  const int lane = tid & 31;
  const int w    = tid >> 5;
  const int wm   = w >> 2;   // 0..1
  const int wn   = w & 3;    // 0..3
  const int Hv   = (DIL == 2) ? 2 * Hin - 1 : Hin;
  const int Wv   = (DIL == 2) ? 2 * Win - 1 : Win;
  const int HWo  = Hout * Wout;

  // staging geometry: this thread always stages k-col kk, rows (tid>>5)+8j
  const int kk   = tid & 31;
  const int pdst = wmma_perm(kk);
  int nj[16], ohj[16], owj[16];
  #pragma unroll
  for (int j = 0; j < 16; ++j) {
    int m  = m0 + (tid >> 5) + 8 * j;
    int n  = m / HWo;
    int hw = m - n * HWo;
    int oh = hw / Wout;
    nj[j] = n; ohj[j] = oh; owj[j] = hw - oh * Wout;
  }

  auto fetchA = [&](int k0, float* v) {
    int  kg  = k0 + kk;
    int  c   = kg % CIN;
    int  rem = kg / CIN;
    int  kw_ = rem % KW;
    int  kh_ = rem / KW;
    bool kok = (kg < KREAL);
    #pragma unroll
    for (int j = 0; j < 16; ++j) {
      float val = 0.f;
      if (kok) {
        int vh = ohj[j] * STRIDE - PAD + kh_;
        int vw = owj[j] * STRIDE - PAD + kw_;
        if (vh >= 0 && vh < Hv && vw >= 0 && vw < Wv) {
          if (DIL == 1)
            val = in[((size_t)(nj[j] * Hin + vh) * Win + vw) * CIN + c];
          else if (((vh | vw) & 1) == 0)
            val = in[((size_t)(nj[j] * Hin + (vh >> 1)) * Win + (vw >> 1)) * CIN + c];
        }
        if (relu_in) val = fmaxf(val, 0.f);
      }
      v[j] = val;
    }
  };
  auto storeA = [&](const float* v) {
    #pragma unroll
    for (int j = 0; j < 16; ++j)
      As[(tid >> 5) + 8 * j][pdst] = (__bf16)v[j];
  };

  CFrag acc[4][2];
  #pragma unroll
  for (int i = 0; i < 4; ++i)
    #pragma unroll
    for (int j = 0; j < 2; ++j)
      #pragma unroll
      for (int q = 0; q < 8; ++q) acc[i][j].e[q] = 0.f;

  const int nsteps = Kpad >> 5;
  float avr[16];

  // prologue: tile 0  (B block s: 128 cols x 32 halves, pad 64B rows by 32B)
  fetchA(0, avr);
  storeA(avr);
  if (tid < 32) {
    tdm_load_2d_bf16((uint32_t)(uintptr_t)&Bsw[0][0][0],
                     Bw + (size_t)n0 * 32, 128, 32, 32, 3, 7);
    __builtin_amdgcn_s_wait_tensorcnt(0);
  }
  __syncthreads();

  for (int s = 0; s < nsteps; ++s) {
    const bool more = (s + 1) < nsteps;
    if (more) {
      fetchA((s + 1) << 5, avr);                       // overlaps WMMA below
      if (tid < 32)
        tdm_load_2d_bf16((uint32_t)(uintptr_t)&Bsw[(s + 1) & 1][0][0],
                         Bw + ((size_t)(s + 1) * Npad + n0) * 32,
                         128, 32, 32, 3, 7);
    }
    // ---- compute step s ----
    AFrag afr[4], bfr[2];
    const int rbase   = wm * 64 + (lane & 15);
    const int cbase   = wn * 32 + (lane & 15);
    const int khalf16 = (lane >> 4) * 16;
    #pragma unroll
    for (int mt = 0; mt < 4; ++mt)
      afr[mt].v = *(const v16bf*)&As[rbase + mt * 16][khalf16];
    #pragma unroll
    for (int nt = 0; nt < 2; ++nt)
      bfr[nt].v = *(const v16bf*)&Bsw[s & 1][cbase + nt * 16][khalf16];
    #pragma unroll
    for (int mt = 0; mt < 4; ++mt)
      #pragma unroll
      for (int nt = 0; nt < 2; ++nt)
        acc[mt][nt].v = __builtin_amdgcn_wmma_f32_16x16x32_bf16(
            false, afr[mt].v, false, bfr[nt].v, (short)0, acc[mt][nt].v,
            false, false);
    __syncthreads();             // everyone done reading As / Bsw[s&1]
    if (more) {
      storeA(avr);               // refill single-buffered As
      if (tid < 32) __builtin_amdgcn_s_wait_tensorcnt(0);  // B(s+1) landed
    }
    __syncthreads();
  }

  // ---- epilogue ----
  #pragma unroll
  for (int mt = 0; mt < 4; ++mt) {
    int rb = m0 + wm * 64 + mt * 16 + (lane >> 4) * 8;
    #pragma unroll
    for (int nt = 0; nt < 2; ++nt) {
      int col = n0 + wn * 32 + nt * 16 + (lane & 15);
      if (col < Cout) {
        float bv = bias ? bias[col] : 0.f;
        #pragma unroll
        for (int i = 0; i < 8; ++i)
          out[(size_t)(rb + i) * Cout + col] = acc[mt][nt].e[i] + bv;
      }
    }
  }
}

// ---------------------------------------------------------------------------
// VQ: per 128-row tile, argmin_k(||w_k||^2 - 2 z.w_k). K=D=128, N chunks of
// 64 with TDM double-buffered pre-swizzled codebook tiles.
// ---------------------------------------------------------------------------
__global__ __launch_bounds__(TB) void vq_argmin(
    const float* __restrict__ zf, const bf16_t* __restrict__ embT,
    const float* __restrict__ wnorm, int* __restrict__ idx_out,
    float* __restrict__ idxf_out)
{
  __shared__ __align__(64) __bf16 Az[128][144];
  __shared__ __align__(64) __bf16 Bzw[2][64][144];   // swizzled codebook tiles
  __shared__ float  Sc[128][65];

  const int tid  = threadIdx.x;
  const int m0   = blockIdx.x * 128;
  const int lane = tid & 31;
  const int w    = tid >> 5;
  const int wm   = w >> 1;   // 0..3
  const int wn   = w & 1;    // 0..1

  const int dz = tid & 127;
  const int pz = ((dz >> 5) << 5) + wmma_perm(dz & 31);
  #pragma unroll 4
  for (int j = 0; j < 64; ++j) {
    int r = 2 * j + (tid >> 7);
    Az[r][pz] = (__bf16)zf[(size_t)(m0 + r) * 128 + dz];
  }
  if (tid < 32) {
    // chunk block: 64 cols x 128 halves (4 k-blocks), pad 256B rows by 32B
    tdm_load_2d_bf16((uint32_t)(uintptr_t)&Bzw[0][0][0],
                     embT, 64, 128, 128, 5, 7);
    __builtin_amdgcn_s_wait_tensorcnt(0);
  }
  __syncthreads();

  float best = 3.4e38f;
  int   bidx = 0;

  for (int ch = 0; ch < 16; ++ch) {
    const int kc = ch * 64;
    if (ch + 1 < 16 && tid < 32)
      tdm_load_2d_bf16((uint32_t)(uintptr_t)&Bzw[(ch + 1) & 1][0][0],
                       embT + (size_t)(ch + 1) * 8192, 64, 128, 128, 5, 7);

    CFrag acc[2][2];
    #pragma unroll
    for (int i = 0; i < 2; ++i)
      #pragma unroll
      for (int j = 0; j < 2; ++j)
        #pragma unroll
        for (int q = 0; q < 8; ++q) acc[i][j].e[q] = 0.f;

    const int rbase   = wm * 32 + (lane & 15);
    const int cbase   = wn * 32 + (lane & 15);
    const int khalf16 = (lane >> 4) * 16;
    #pragma unroll
    for (int k0 = 0; k0 < 128; k0 += 32) {
      AFrag afr[2], bfr[2];
      #pragma unroll
      for (int mt = 0; mt < 2; ++mt)
        afr[mt].v = *(const v16bf*)&Az[rbase + mt * 16][k0 + khalf16];
      #pragma unroll
      for (int nt = 0; nt < 2; ++nt)
        bfr[nt].v = *(const v16bf*)&Bzw[ch & 1][cbase + nt * 16][k0 + khalf16];
      #pragma unroll
      for (int mt = 0; mt < 2; ++mt)
        #pragma unroll
        for (int nt = 0; nt < 2; ++nt)
          acc[mt][nt].v = __builtin_amdgcn_wmma_f32_16x16x32_bf16(
              false, afr[mt].v, false, bfr[nt].v, (short)0, acc[mt][nt].v,
              false, false);
    }

    #pragma unroll
    for (int mt = 0; mt < 2; ++mt) {
      int rb = wm * 32 + mt * 16 + (lane >> 4) * 8;
      #pragma unroll
      for (int nt = 0; nt < 2; ++nt) {
        int col = wn * 32 + nt * 16 + (lane & 15);
        #pragma unroll
        for (int i = 0; i < 8; ++i)
          Sc[rb + i][col] = -2.f * acc[mt][nt].e[i];
      }
    }
    __syncthreads();
    if (tid < 128) {
      for (int c = 0; c < 64; ++c) {
        float s = Sc[tid][c] + wnorm[kc + c];
        if (s < best) { best = s; bidx = kc + c; }
      }
    }
    if (tid < 32) __builtin_amdgcn_s_wait_tensorcnt(0);
    __syncthreads();
  }
  if (tid < 128) {
    idx_out[m0 + tid]  = bidx;
    idxf_out[m0 + tid] = (float)bidx;
  }
}

// -------------------------- elementwise / reductions ------------------------
__global__ void zero_f32(float* p, int n) {
  int i = blockIdx.x * TB + threadIdx.x;
  if (i < n) p[i] = 0.f;
}

__global__ void bn_stats(const float* __restrict__ x, int M, float* __restrict__ stats) {
  __shared__ float ssum[TB], ssq[TB];
  int c = threadIdx.x & 127;
  float s = 0.f, q = 0.f;
  for (int r = blockIdx.x * 2 + (threadIdx.x >> 7); r < M; r += gridDim.x * 2) {
    float v = x[(size_t)r * 128 + c];
    s += v; q += v * v;
  }
  ssum[threadIdx.x] = s; ssq[threadIdx.x] = q;
  __syncthreads();
  if (threadIdx.x < 128) {
    atomicAdd(&stats[c],       ssum[threadIdx.x] + ssum[threadIdx.x + 128]);
    atomicAdd(&stats[128 + c], ssq[threadIdx.x]  + ssq[threadIdx.x + 128]);
  }
}

__global__ void bn_finalize(const float* stats, const float* g, const float* b,
                            float* ss, float invM) {
  int c = threadIdx.x;   // 128
  float mean = stats[c] * invM;
  float var  = stats[128 + c] * invM - mean * mean;
  float sc   = g[c] * rsqrtf(var + 1e-5f);
  ss[c]       = sc;
  ss[128 + c] = b[c] - mean * sc;
}

__global__ void bn_apply(const float* __restrict__ x, const float* __restrict__ ss,
                         const float* __restrict__ res, float* __restrict__ out,
                         int total, int relu) {
  for (int i = blockIdx.x * TB + threadIdx.x; i < total; i += gridDim.x * TB) {
    int c = i & 127;
    float v = x[i] * ss[c] + ss[128 + c];
    if (relu) v = fmaxf(v, 0.f);
    if (res)  v += res[i];
    out[i] = v;
  }
}

__global__ void vq_gather(const float* __restrict__ z, const float* __restrict__ emb,
                          const int* __restrict__ idx, float* __restrict__ qn,
                          float* __restrict__ q_nchw, float* __restrict__ lossacc,
                          int Mrows) {
  __shared__ float red[TB];
  float ls = 0.f;
  int total = Mrows * 128;
  for (int i = blockIdx.x * TB + threadIdx.x; i < total; i += gridDim.x * TB) {
    int m = i >> 7, c = i & 127;
    int k = idx[m];
    float qv = emb[(size_t)k * 128 + c];
    qn[i] = qv;
    float d = qv - z[i];
    ls += d * d;
    int n = m >> 12, hw = m & 4095;
    q_nchw[(((size_t)n * 128 + c) * 64 + (hw >> 6)) * 64 + (hw & 63)] = qv;
  }
  red[threadIdx.x] = ls; __syncthreads();
  for (int s = 128; s > 0; s >>= 1) {
    if (threadIdx.x < s) red[threadIdx.x] += red[threadIdx.x + s];
    __syncthreads();
  }
  if (threadIdx.x == 0) atomicAdd(lossacc, red[0]);
}

__global__ void loss_final(const float* lacc, float* out) {
  out[0] = lacc[0] * 1.25f / (65536.0f * 128.0f);
}

__global__ void nchw_to_nhwc(const float* __restrict__ in, float* __restrict__ out,
                             int Nb, int C, int H, int W) {
  size_t total = (size_t)Nb * C * H * W;
  for (size_t i = blockIdx.x * (size_t)TB + threadIdx.x; i < total; i += (size_t)gridDim.x * TB) {
    int c = (int)(i % C); size_t t = i / C;
    int ww = (int)(t % W); t /= W;
    int h  = (int)(t % H); int n = (int)(t / H);
    out[i] = in[(((size_t)n * C + c) * H + h) * W + ww];
  }
}

__global__ void nhwc_to_nchw(const float* __restrict__ in, float* __restrict__ out,
                             int Nb, int C, int H, int W) {
  size_t total = (size_t)Nb * C * H * W;
  for (size_t i = blockIdx.x * (size_t)TB + threadIdx.x; i < total; i += (size_t)gridDim.x * TB) {
    int ww = (int)(i % W); size_t t = i / W;
    int h  = (int)(t % H); t /= H;
    int c  = (int)(t % C); int n = (int)(t / C);
    out[i] = in[(((size_t)n * H + h) * W + ww) * C + c];
  }
}

// ----------------------------- weight packing -------------------------------
// Packed layout: per 32-k block kb, [Npad cols][32 halves] with fragment-order
// permutation inside the 32 => TDM copies it verbatim into WMMA-ready LDS.
__global__ void pack_conv(const float* __restrict__ Wsrc, bf16_t* __restrict__ Bdst,
                          int O, int I, int KH, int KW, int Kpad, int Npad) {
  int i = blockIdx.x * TB + threadIdx.x;
  if (i >= Kpad * Npad) return;
  int o = i % Npad, k = i / Npad;
  float v = 0.f;
  if (o < O && k < KH * KW * I) {
    int c = k % I, rem = k / I;
    int kw = rem % KW, kh = rem / KW;
    v = Wsrc[(((size_t)o * I + c) * KH + kh) * KW + kw];
  }
  int kb = k >> 5, p = wmma_perm(k & 31);
  Bdst[((size_t)kb * Npad + o) * 32 + p] = (bf16_t)v;
}

__global__ void pack_deconv(const float* __restrict__ Wsrc, bf16_t* __restrict__ Bdst,
                            int I, int O, int KH, int KW, int Kpad, int Npad) {
  int i = blockIdx.x * TB + threadIdx.x;
  if (i >= Kpad * Npad) return;
  int o = i % Npad, k = i / Npad;
  float v = 0.f;
  if (o < O && k < KH * KW * I) {
    int c = k % I, rem = k / I;
    int kw = rem % KW, kh = rem / KW;
    v = Wsrc[(((size_t)c * O + o) * KH + (KH - 1 - kh)) * KW + (KW - 1 - kw)];
  }
  int kb = k >> 5, p = wmma_perm(k & 31);
  Bdst[((size_t)kb * Npad + o) * 32 + p] = (bf16_t)v;
}

// Codebook: chunk ch (64 cols) -> [64 cols][4 kb][32 perm halves] contiguous.
__global__ void pack_embT(const float* __restrict__ emb, bf16_t* __restrict__ embT) {
  int i = blockIdx.x * TB + threadIdx.x;
  if (i >= 1024 * 128) return;
  int k = i >> 7, d = i & 127;
  int ch = k >> 6, col = k & 63, kb = d >> 5, p = wmma_perm(d & 31);
  embT[(((size_t)ch * 64 + col) * 4 + kb) * 32 + p] = (bf16_t)emb[i];
}

__global__ void wnorm_k(const float* __restrict__ emb, float* __restrict__ wn) {
  int k = blockIdx.x * TB + threadIdx.x;
  if (k >= 1024) return;
  float s = 0.f;
  for (int d = 0; d < 128; ++d) { float v = emb[(size_t)k * 128 + d]; s += v * v; }
  wn[k] = s;
}

// ----------------------------- host orchestration ---------------------------
template<int CIN, int KH, int KW, int STRIDE, int PAD, int DIL>
static inline void launch_conv(const float* in, const bf16_t* Bw, const float* bias,
                               float* out, int Nb, int Hin, int Win,
                               int Hout, int Wout, int Cout,
                               int Kpad, int Npad, int relu_in, hipStream_t s) {
  int M = Nb * Hout * Wout;
  dim3 grid(M / 128, Npad / 128);
  conv_gemm<CIN, KH, KW, STRIDE, PAD, DIL><<<grid, TB, 0, s>>>(
      in, Bw, bias, out, Hin, Win, Hout, Wout, Cout, Kpad, Npad, relu_in);
}

static inline void run_bn(const float* act, const float* g, const float* b,
                          const float* res, float* dst, int M, int relu,
                          float* stats, float* ss, hipStream_t s) {
  zero_f32<<<1, TB, 0, s>>>(stats, 256);
  bn_stats<<<128, TB, 0, s>>>(act, M, stats);
  bn_finalize<<<1, 128, 0, s>>>(stats, g, b, ss, 1.f / (float)M);
  bn_apply<<<1024, TB, 0, s>>>(act, ss, res, dst, M * 128, relu);
}

extern "C" void kernel_launch(void* const* d_in, const int* in_sizes, int n_in,
                              void* d_out, int out_size, void* d_ws, size_t ws_size,
                              hipStream_t stream) {
  (void)in_sizes; (void)n_in; (void)out_size; (void)ws_size;
  const float* x = (const float*)d_in[0];
  auto P = [&](int i) { return (const float*)d_in[i]; };
  float* out = (float*)d_out;
  const size_t OFF_Y = 0, OFF_X = 3145728, OFF_Q = 6291456,
               OFF_I = 14680064, OFF_L = 14745600;

  // workspace layout (floats)
  float* ws = (float*)d_ws;
  size_t o = 0;
  float* actA = ws + o; o += 33554432;         // 16x128x128x128
  float* actZ = ws + o; o += 8388608;          // 16x64x64x128
  float* tmp1 = ws + o; o += 8388608;
  float* tmp2 = ws + o; o += 8388608;
  float* qbuf = ws + o; o += 8388608;
  float* xnh  = ws + o; o += 3145728;          // 16x256x256x3
  float* ynh  = ws + o; o += 3145728;
  float* statsb = ws + o; o += 256;
  float* ssb    = ws + o; o += 256;
  float* wnb    = ws + o; o += 1024;
  float* lacc   = ws + o; o += 4;
  int*   idxb   = (int*)(ws + o); o += 65536;
  bf16_t* pw = (bf16_t*)(ws + o);
  size_t po = 0;
  bf16_t* pw_e1 = pw + po; po += 64 * 128;
  bf16_t* pw_e2 = pw + po; po += 2048 * 128;
  bf16_t* pw_rc1[6]; bf16_t* pw_rc2[6];
  for (int r = 0; r < 6; ++r) {
    pw_rc1[r] = pw + po; po += 1152 * 128;
    pw_rc2[r] = pw + po; po += 128 * 128;
  }
  bf16_t* pw_d1 = pw + po; po += 2048 * 128;
  bf16_t* pw_d2 = pw + po; po += 2048 * 128;
  bf16_t* embT  = pw + po; po += 128 * 1024;

  auto cdiv = [](size_t a, size_t b) { return (unsigned)((a + b - 1) / b); };

  // ---- pack weights (bf16, fragment-order) ----
  pack_conv<<<cdiv(64 * 128, TB), TB, 0, stream>>>(P(1), pw_e1, 128, 3, 4, 4, 64, 128);
  pack_conv<<<cdiv(2048 * 128, TB), TB, 0, stream>>>(P(5), pw_e2, 128, 128, 4, 4, 2048, 128);
  for (int r = 0; r < 6; ++r) {
    int base = 9 + 6 * r;
    pack_conv<<<cdiv(1152 * 128, TB), TB, 0, stream>>>(P(base),     pw_rc1[r], 128, 128, 3, 3, 1152, 128);
    pack_conv<<<cdiv(128 * 128, TB),  TB, 0, stream>>>(P(base + 3), pw_rc2[r], 128, 128, 1, 1, 128, 128);
  }
  pack_deconv<<<cdiv(2048 * 128, TB), TB, 0, stream>>>(P(45), pw_d1, 128, 128, 4, 4, 2048, 128);
  pack_deconv<<<cdiv(2048 * 128, TB), TB, 0, stream>>>(P(49), pw_d2, 128, 3, 4, 4, 2048, 128);
  pack_embT<<<cdiv(131072, TB), TB, 0, stream>>>(P(51), embT);
  wnorm_k<<<cdiv(1024, TB), TB, 0, stream>>>(P(51), wnb);

  // ---- encoder ----
  nchw_to_nhwc<<<12288, TB, 0, stream>>>(x, xnh, 16, 3, 256, 256);
  launch_conv<3, 4, 4, 2, 1, 1>(xnh, pw_e1, P(2), actA, 16, 256, 256, 128, 128, 128, 64, 128, 0, stream);
  run_bn(actA, P(3), P(4), nullptr, actA, 262144, 1, statsb, ssb, stream);
  launch_conv<128, 4, 4, 2, 1, 1>(actA, pw_e2, P(6), actZ, 16, 128, 128, 64, 64, 128, 2048, 128, 0, stream);
  run_bn(actZ, P(7), P(8), nullptr, actZ, 65536, 0, statsb, ssb, stream);
  for (int r = 0; r < 3; ++r) {
    int base = 9 + 6 * r;
    launch_conv<128, 3, 3, 1, 1, 1>(actZ, pw_rc1[r], nullptr, tmp1, 16, 64, 64, 64, 64, 128, 1152, 128, 1, stream);
    run_bn(tmp1, P(base + 1), P(base + 2), nullptr, tmp1, 65536, 0, statsb, ssb, stream);
    launch_conv<128, 1, 1, 1, 0, 1>(tmp1, pw_rc2[r], nullptr, tmp2, 16, 64, 64, 64, 64, 128, 128, 128, 1, stream);
    run_bn(tmp2, P(base + 4), P(base + 5), actZ, actZ, 65536, 0, statsb, ssb, stream);
  }

  // ---- VQ ----
  vq_argmin<<<512, TB, 0, stream>>>(actZ, embT, wnb, idxb, out + OFF_I);
  zero_f32<<<1, TB, 0, stream>>>(lacc, 1);
  vq_gather<<<2048, TB, 0, stream>>>(actZ, P(51), idxb, qbuf, out + OFF_Q, lacc, 65536);
  loss_final<<<1, 1, 0, stream>>>(lacc, out + OFF_L);

  // ---- decoder ----
  for (int r = 3; r < 6; ++r) {
    int base = 9 + 6 * r;
    launch_conv<128, 3, 3, 1, 1, 1>(qbuf, pw_rc1[r], nullptr, tmp1, 16, 64, 64, 64, 64, 128, 1152, 128, 1, stream);
    run_bn(tmp1, P(base + 1), P(base + 2), nullptr, tmp1, 65536, 0, statsb, ssb, stream);
    launch_conv<128, 1, 1, 1, 0, 1>(tmp1, pw_rc2[r], nullptr, tmp2, 16, 64, 64, 64, 64, 128, 128, 128, 1, stream);
    run_bn(tmp2, P(base + 4), P(base + 5), qbuf, qbuf, 65536, 0, statsb, ssb, stream);
  }
  // relu folded into deconv1's A-loader; DIL=2 => ConvTranspose2d(4,2,1)
  launch_conv<128, 4, 4, 1, 2, 2>(qbuf, pw_d1, P(46), actA, 16, 64, 64, 128, 128, 128, 2048, 128, 1, stream);
  run_bn(actA, P(47), P(48), nullptr, actA, 262144, 1, statsb, ssb, stream);
  launch_conv<128, 4, 4, 1, 2, 2>(actA, pw_d2, P(50), ynh, 16, 128, 128, 256, 256, 3, 2048, 128, 0, stream);
  nhwc_to_nchw<<<12288, TB, 0, stream>>>(ynh, out + OFF_Y, 16, 3, 256, 256);

  // ---- passthrough x ----
  hipMemcpyAsync(out + OFF_X, x, 3145728 * sizeof(float),
                 hipMemcpyDeviceToDevice, stream);
}
